// PSALayer_19456201851415
// MI455X (gfx1250) — compile-verified, matrix operand order checked by reading, not run on hardware
//
#include <hip/hip_runtime.h>
#include <hip/hip_bf16.h>
#include <math.h>

// Problem constants (match reference)
#define Bsz 1024
#define Nn  16384
#define Dd  512
#define Kk  10
#define KSTEPS (Dd / 32)       // 16 K-steps of 32

typedef __attribute__((ext_vector_type(16))) __bf16 v16bf;
typedef __attribute__((ext_vector_type(8)))  float  v8f;

// ---------------------------------------------------------------------------
// Fragment gather from row-major f32 (used by prep kernel and fallback GEMM).
// Per ISA 7.12.2 (16-bit A 16x32 / B 32x16 layouts), a lane holds two
// contiguous K-octets: [base .. base+7] and [base+16 .. base+23], where
// base = k0 + (lane>=16 ? 8 : 0).
// ---------------------------------------------------------------------------
__device__ __forceinline__ v16bf load_frag_bf16(const float* __restrict__ row, int base) {
    const float4* p0 = (const float4*)(row + base);
    float4 a = p0[0];
    float4 b = p0[1];
    const float4* p1 = (const float4*)(row + base + 16);
    float4 c = p1[0];
    float4 d = p1[1];
    v16bf r;
    r[0]  = (__bf16)a.x; r[1]  = (__bf16)a.y; r[2]  = (__bf16)a.z; r[3]  = (__bf16)a.w;
    r[4]  = (__bf16)b.x; r[5]  = (__bf16)b.y; r[6]  = (__bf16)b.z; r[7]  = (__bf16)b.w;
    r[8]  = (__bf16)c.x; r[9]  = (__bf16)c.y; r[10] = (__bf16)c.z; r[11] = (__bf16)c.w;
    r[12] = (__bf16)d.x; r[13] = (__bf16)d.y; r[14] = (__bf16)d.z; r[15] = (__bf16)d.w;
    return r;
}

// ---------------------------------------------------------------------------
// Prep: convert f32 row-major [rows x 512] -> bf16, swizzled so each WMMA
// fragment is contiguous: frag = rt*KSTEPS + ks holds 32 lanes x 16 bf16
// at dst[frag*512 + lane*16]. One wave per fragment.
// ---------------------------------------------------------------------------
__global__ __launch_bounds__(256) void psa_convert_swizzle(
    const float* __restrict__ src, __bf16* __restrict__ dst)
{
    const int lane = threadIdx.x & 31;
    const int frag = blockIdx.x * (blockDim.x >> 5) + (threadIdx.x >> 5);
    const int rt = frag >> 4;             // 16-row tile index
    const int ks = frag & (KSTEPS - 1);   // 32-wide K step
    const int row = rt * 16 + (lane & 15);
    const int base = ks * 32 + ((lane >> 4) << 3);
    v16bf v = load_frag_bf16(src + (size_t)row * Dd, base);
    *(v16bf*)(dst + (size_t)frag * 512 + (size_t)lane * 16) = v;   // 1KB/instr, coalesced
}

// ---------------------------------------------------------------------------
// GEMM (swizzled bf16 operands): cur[b,n] = sum_d x[b,d] * w[n,d]
// One wave computes a 32x64 output tile: 2 (M) x 4 (N) accumulators.
// Inner loop: 12 global_load_b128 + 8 v_wmma_f32_16x16x32_bf16 per K-step.
// ---------------------------------------------------------------------------
__global__ __launch_bounds__(256) void psa_gemm_wmma_swz(
    const __bf16* __restrict__ xs,   // swizzled [ (B/16)*16 frags ][512]
    const __bf16* __restrict__ wsz,  // swizzled [ (N/16)*16 frags ][512]
    float*        __restrict__ cur)  // [B, N]
{
    const int lane = threadIdx.x & 31;
    const int wid  = blockIdx.x * (blockDim.x >> 5) + (threadIdx.x >> 5);
    const int wavesN = Nn / 64;              // 256
    const int wn = wid % wavesN;
    const int wm = wid / wavesN;             // 0..31
    const int m0 = wm * 32;
    const int n0 = wn * 64;

    // Fragment base pointers (advance by 512 bf16 per K-step).
    const size_t lofs = (size_t)lane * 16;
    const __bf16* ap0 = xs  + ((size_t)(wm * 2 + 0) * KSTEPS) * 512 + lofs;
    const __bf16* ap1 = xs  + ((size_t)(wm * 2 + 1) * KSTEPS) * 512 + lofs;
    const __bf16* bp[4];
#pragma unroll
    for (int j = 0; j < 4; ++j)
        bp[j] = wsz + ((size_t)(wn * 4 + j) * KSTEPS) * 512 + lofs;

    v8f acc[2][4];
#pragma unroll
    for (int i = 0; i < 2; ++i)
#pragma unroll
        for (int j = 0; j < 4; ++j)
            acc[i][j] = (v8f){0.f, 0.f, 0.f, 0.f, 0.f, 0.f, 0.f, 0.f};

    for (int ks = 0; ks < KSTEPS; ++ks) {
        const size_t o = (size_t)ks * 512;
        v16bf af0 = *(const v16bf*)(ap0 + o);
        v16bf af1 = *(const v16bf*)(ap1 + o);
        v16bf bf[4];
#pragma unroll
        for (int j = 0; j < 4; ++j)
            bf[j] = *(const v16bf*)(bp[j] + o);

#pragma unroll
        for (int j = 0; j < 4; ++j) {
            acc[0][j] = __builtin_amdgcn_wmma_f32_16x16x32_bf16(
                false, af0, false, bf[j], (short)0, acc[0][j], false, false);
            acc[1][j] = __builtin_amdgcn_wmma_f32_16x16x32_bf16(
                false, af1, false, bf[j], (short)0, acc[1][j], false, false);
        }
    }

    // C/D layout (16x16 f32): VGPR r -> M = r (lanes 0-15) or 8+r (lanes 16-31);
    // N = lane % 16.
    const int lrow = lane & 15;
    const int mo   = (lane >> 4) << 3;
#pragma unroll
    for (int i = 0; i < 2; ++i) {
#pragma unroll
        for (int j = 0; j < 4; ++j) {
            const int col = n0 + j * 16 + lrow;
#pragma unroll
            for (int r = 0; r < 8; ++r) {
                const int row = m0 + i * 16 + mo + r;
                cur[(size_t)row * Nn + col] = acc[i][j][r];
            }
        }
    }
}

// ---------------------------------------------------------------------------
// Fallback GEMM (direct f32 loads + in-loop conversion) when d_ws is too
// small for the bf16 operand staging.
// ---------------------------------------------------------------------------
__global__ __launch_bounds__(256) void psa_gemm_wmma_direct(
    const float* __restrict__ x,     // [B, D]
    const float* __restrict__ w,     // [N, D]
    float*       __restrict__ cur)   // [B, N]
{
    const int lane  = threadIdx.x & 31;
    const int wid   = blockIdx.x * (blockDim.x >> 5) + (threadIdx.x >> 5);
    const int wavesN = Nn / 64;
    const int wn = wid % wavesN;
    const int wm = wid / wavesN;
    const int m0 = wm * 32;
    const int n0 = wn * 64;
    const int lrow  = lane & 15;
    const int khalf = (lane >> 4) << 3;

    const float* arow0 = x + (size_t)(m0 + lrow) * Dd;
    const float* arow1 = x + (size_t)(m0 + 16 + lrow) * Dd;
    const float* brow[4];
#pragma unroll
    for (int j = 0; j < 4; ++j)
        brow[j] = w + (size_t)(n0 + j * 16 + lrow) * Dd;

    v8f acc[2][4];
#pragma unroll
    for (int i = 0; i < 2; ++i)
#pragma unroll
        for (int j = 0; j < 4; ++j)
            acc[i][j] = (v8f){0.f, 0.f, 0.f, 0.f, 0.f, 0.f, 0.f, 0.f};

    for (int k0 = 0; k0 < Dd; k0 += 32) {
        const int base = k0 + khalf;
        v16bf af0 = load_frag_bf16(arow0, base);
        v16bf af1 = load_frag_bf16(arow1, base);
        v16bf bf[4];
#pragma unroll
        for (int j = 0; j < 4; ++j)
            bf[j] = load_frag_bf16(brow[j], base);

#pragma unroll
        for (int j = 0; j < 4; ++j) {
            acc[0][j] = __builtin_amdgcn_wmma_f32_16x16x32_bf16(
                false, af0, false, bf[j], (short)0, acc[0][j], false, false);
            acc[1][j] = __builtin_amdgcn_wmma_f32_16x16x32_bf16(
                false, af1, false, bf[j], (short)0, acc[1][j], false, false);
        }
    }

    const int mo = (lane >> 4) << 3;
#pragma unroll
    for (int i = 0; i < 2; ++i) {
#pragma unroll
        for (int j = 0; j < 4; ++j) {
            const int col = n0 + j * 16 + lrow;
#pragma unroll
            for (int r = 0; r < 8; ++r) {
                const int row = m0 + i * 16 + mo + r;
                cur[(size_t)row * Nn + col] = acc[i][j][r];
            }
        }
    }
}

// ---------------------------------------------------------------------------
// LIF scan over the batch/time dim (in-place currents -> spikes),
// plus rate + prediction epilogue. One thread per neuron.
// ---------------------------------------------------------------------------
__global__ __launch_bounds__(256) void psa_lif_pred(
    float*       __restrict__ cur,     // [B, N] in: currents, out: spikes
    const float* __restrict__ fastw,   // [N, K]
    const float* __restrict__ sloww,   // [N, K]
    const float* __restrict__ bias,    // [N, K]
    const float* __restrict__ mem0,    // [N]
    float*       __restrict__ pred)    // [N, K]
{
    const int n = blockIdx.x * blockDim.x + threadIdx.x;

    float m = mem0[n];
    float h[5] = {0.f, 0.f, 0.f, 0.f, 0.f};   // ring of last 5 spikes (B >= W)

    int b = 0;
    for (; b + 5 <= Bsz; b += 5) {
#pragma unroll
        for (int j = 0; j < 5; ++j) {
            const size_t idx = (size_t)(b + j) * Nn + n;
            const float c = cur[idx];
            m = 0.9f * m + c;
            const float s = (m >= 1.0f) ? 1.0f : 0.0f;
            m -= s;
            cur[idx] = s;
            h[j] = s;
        }
    }
#pragma unroll
    for (int j = 0; j < 4; ++j) {             // tail: 1024 = 5*204 + 4
        if (b + j < Bsz) {
            const size_t idx = (size_t)(b + j) * Nn + n;
            const float c = cur[idx];
            m = 0.9f * m + c;
            const float s = (m >= 1.0f) ? 1.0f : 0.0f;
            m -= s;
            cur[idx] = s;
            h[j] = s;
        }
    }

    const float rate = (h[0] + h[1] + h[2] + h[3] + h[4]) * 0.2f;

#pragma unroll
    for (int k = 0; k < Kk; ++k) {
        const int e = n * Kk + k;
        const float z = (fastw[e] + sloww[e]) * rate + bias[e];
        pred[e] = 1.0f / (1.0f + __expf(-z));
    }
}

extern "C" void kernel_launch(void* const* d_in, const int* in_sizes, int n_in,
                              void* d_out, int out_size, void* d_ws, size_t ws_size,
                              hipStream_t stream) {
    (void)in_sizes; (void)n_in; (void)out_size;

    const float* x   = (const float*)d_in[0];  // [B, D]
    const float* iw  = (const float*)d_in[1];  // [N, D]
    const float* fw  = (const float*)d_in[2];  // [N, K]
    const float* sw  = (const float*)d_in[3];  // [N, K]
    const float* pb  = (const float*)d_in[4];  // [N, K]
    const float* m0  = (const float*)d_in[5];  // [N]
    // d_in[6] = spike_hist0 [N, W]: irrelevant since B >= W (fully rolled out)

    float* out  = (float*)d_out;
    float* cur  = out;                          // spikes region doubles as currents
    float* pred = out + (size_t)Bsz * Nn;

    const size_t xbytes = (size_t)Bsz * Dd * sizeof(__bf16);   // 1 MB
    const size_t wbytes = (size_t)Nn  * Dd * sizeof(__bf16);   // 16 MB

    if (d_ws && ws_size >= xbytes + wbytes) {
        __bf16* xs  = (__bf16*)d_ws;
        __bf16* wsz = (__bf16*)((char*)d_ws + xbytes);
        // x: (1024/16)*16 = 1024 frags, 8 frags/block -> 128 blocks
        psa_convert_swizzle<<<(Bsz / 16) * KSTEPS / 8, 256, 0, stream>>>(x, xs);
        // W: (16384/16)*16 = 16384 frags -> 2048 blocks
        psa_convert_swizzle<<<(Nn / 16) * KSTEPS / 8, 256, 0, stream>>>(iw, wsz);
        // GEMM: 32 (M strips) * 256 (N strips) = 8192 waves, 8 waves/block.
        psa_gemm_wmma_swz<<<1024, 256, 0, stream>>>(xs, wsz, cur);
    } else {
        psa_gemm_wmma_direct<<<1024, 256, 0, stream>>>(x, iw, cur);
    }

    // LIF + predictions: one thread per neuron.
    psa_lif_pred<<<Nn / 256, 256, 0, stream>>>(cur, fw, sw, pb, m0, pred);
}
